// MemModule_72567767433400
// MI455X (gfx1250) — matrix operand order repaired and checked by reading, not compiled, not run on hardware
//
#include <hip/hip_runtime.h>
#include <hip/hip_bf16.h>

// ---------------------------------------------------------------------------
// MemModule attention, fused for MI455X (gfx1250, wave32, WMMA bf16)
//   w   = z @ mem^T            [16384 x 4096]
//   w   = softmax(w, axis=1)
//   h   = hardshrink-relu(w)
//   att = softmax(h, axis=1)   -> output[0]  (f32)
//   zh  = att @ mem            [16384 x 1024]  -> output[1]  (f32)
//
// TILE_B = 32 batch rows per WG; scores kept as bf16 in LDS (f32 math in
// registers). Each wave loads one B-fragment and feeds TWO wmma (both 16-row
// halves), halving L2 traffic for the streamed memory operand. Softmax sweeps
// are 8-wide vectorized (ds_load_b128 / global_store_b128).
// ---------------------------------------------------------------------------

typedef __bf16 bf16_t;
typedef __attribute__((ext_vector_type(16))) __bf16 v16bf;
typedef __attribute__((ext_vector_type(8)))  __bf16 v8bf;
typedef __attribute__((ext_vector_type(8)))  float  v8f;

constexpr int MEM_DIM = 4096;
constexpr int FEAT    = 1024;
constexpr int BATCH   = 16384;
constexpr int TILE_B  = 32;
constexpr int THREADS = 512;      // 16 waves
constexpr int NWAVES  = THREADS / 32;
constexpr float EPS   = 1e-12f;

// LDS: [ z tile bf16: 32*1024*2 = 64KB ][ scores bf16: 32*4096*2 = 256KB ]
constexpr unsigned SMEM_BYTES = TILE_B * FEAT * sizeof(bf16_t)
                              + TILE_B * MEM_DIM * sizeof(bf16_t); // 327680 B

__device__ __forceinline__ float wave_max(float v) {
#pragma unroll
  for (int off = 16; off > 0; off >>= 1) v = fmaxf(v, __shfl_xor(v, off, 32));
  return v;
}
__device__ __forceinline__ float wave_sum(float v) {
#pragma unroll
  for (int off = 16; off > 0; off >>= 1) v += __shfl_xor(v, off, 32);
  return v;
}

// A-fragment (16-bit, 16x32): per lane two contiguous 8-element chunks,
// K = {koff..koff+7} and {koff+16..koff+23}, koff = 0 (lanes 0-15) / 8 (16-31)
__device__ __forceinline__ v16bf load_afrag(const bf16_t* p) {
  v8bf lo = *(const v8bf*)(p);
  v8bf hi = *(const v8bf*)(p + 16);
  return __builtin_shufflevector(lo, hi, 0, 1, 2, 3, 4, 5, 6, 7,
                                         8, 9, 10, 11, 12, 13, 14, 15);
}

// --- prep: memory f32 -> bf16 row-major copy (GEMM1 B) + transposed copy (GEMM2 B)
__global__ void __launch_bounds__(256)
prep_mem_kernel(const float* __restrict__ mem, bf16_t* __restrict__ memB,
                bf16_t* __restrict__ memT) {
  const int m = blockIdx.x;                       // 0..4095
  for (int c = threadIdx.x; c < FEAT; c += blockDim.x) {
    float v = mem[(size_t)m * FEAT + c];
    bf16_t b = (bf16_t)v;
    memB[(size_t)m * FEAT + c] = b;               // [M, F] row-major
    memT[(size_t)c * MEM_DIM + m] = b;            // [F, M] = mem^T row-major
  }
}

__global__ void __launch_bounds__(THREADS)
mem_attn_kernel(const float* __restrict__ z, const bf16_t* __restrict__ memB,
                const bf16_t* __restrict__ memT, float* __restrict__ att_out,
                float* __restrict__ zhat_out) {
  extern __shared__ __align__(128) char smem[];
  bf16_t* zs  = (bf16_t*)smem;                                    // 32 x 1024
  bf16_t* wbf = (bf16_t*)(smem + TILE_B * FEAT * sizeof(bf16_t)); // 32 x 4096

  const int tid  = threadIdx.x;
  const int lane = tid & 31;
  const int wid  = tid >> 5;          // 0..15
  const int ln   = lane & 15;         // fragment row/col index
  const int half = lane >> 4;         // lane half (K split)
  const int koffA = half ? 8 : 0;     // A-frag: K {0..7,16..23} / {8..15,24..31}
  const int koffB = half ? 16 : 0;    // unused; B-frag also loads via koffA (same K order as A)
  const int rb = blockIdx.x * TILE_B; // batch row base

  // ---- stage z tile into LDS as bf16 (coalesced) ----
  for (int i = tid; i < TILE_B * FEAT; i += THREADS)
    zs[i] = (bf16_t)z[(size_t)rb * FEAT + i];
  __syncthreads();

  // ---- GEMM1: w[32 x 4096] = z_tile @ mem^T ----
  // One B-fragment per k-step feeds both 16-row halves (2 wmma / B-load).
  {
    const bf16_t* arow0 = zs + ln * FEAT;          // rows 0..15
    const bf16_t* arow1 = zs + (16 + ln) * FEAT;   // rows 16..31
    for (int ct = wid; ct < MEM_DIM / 16; ct += NWAVES) {
      const int mbase = ct * 16;
      const bf16_t* brow = memB + (size_t)(mbase + ln) * FEAT;
      v8f acc0 = {0.f, 0.f, 0.f, 0.f, 0.f, 0.f, 0.f, 0.f};
      v8f acc1 = {0.f, 0.f, 0.f, 0.f, 0.f, 0.f, 0.f, 0.f};
#pragma unroll 4
      for (int k = 0; k < FEAT; k += 32) {
        v16bf b  = load_afrag(brow + k + koffA);
        v16bf a0 = load_afrag(arow0 + k + koffA);
        v16bf a1 = load_afrag(arow1 + k + koffA);
        __builtin_prefetch(brow + k + 256, 0, 3);  // near-cache prefetch
        acc0 = __builtin_amdgcn_wmma_f32_16x16x32_bf16(false, a0, false, b,
                                                       (short)0, acc0, false, false);
        acc1 = __builtin_amdgcn_wmma_f32_16x16x32_bf16(false, a1, false, b,
                                                       (short)0, acc1, false, false);
      }
#pragma unroll
      for (int v = 0; v < 8; ++v) {
        wbf[(v + 8 * half) * MEM_DIM + mbase + ln]      = (bf16_t)acc0[v];
        wbf[(16 + v + 8 * half) * MEM_DIM + mbase + ln] = (bf16_t)acc1[v];
      }
    }
  }
  __syncthreads();

  // ---- softmax -> hardshrink-relu -> softmax; 2 rows/wave, 8-wide sweeps ----
  const float lambd = 2.0f / (float)MEM_DIM;
  for (int r = 2 * wid; r < 2 * wid + 2; ++r) {
    bf16_t* row = wbf + r * MEM_DIM;

    float m1 = -3.4e38f;
    for (int i = lane * 8; i < MEM_DIM; i += 256) {
      v8f x = __builtin_convertvector(*(const v8bf*)(row + i), v8f);
#pragma unroll
      for (int j = 0; j < 8; ++j) m1 = fmaxf(m1, x[j]);
    }
    m1 = wave_max(m1);

    float s1 = 0.f;
    for (int i = lane * 8; i < MEM_DIM; i += 256) {
      v8f x = __builtin_convertvector(*(const v8bf*)(row + i), v8f);
#pragma unroll
      for (int j = 0; j < 8; ++j) s1 += __expf(x[j] - m1);
    }
    s1 = wave_sum(s1);
    const float inv1 = 1.0f / s1;

    float m2 = 0.f;   // hardshrink output is >= 0
    for (int i = lane * 8; i < MEM_DIM; i += 256) {
      v8f x = __builtin_convertvector(*(const v8bf*)(row + i), v8f);
      v8f h;
#pragma unroll
      for (int j = 0; j < 8; ++j) {
        float p = __expf(x[j] - m1) * inv1;               // softmax1
        float d = p - lambd;
        float hv = fmaxf(d, 0.f) * p / (fabsf(d) + EPS);  // hardshrink-relu
        h[j] = hv;
        m2 = fmaxf(m2, hv);
      }
      *(v8bf*)(row + i) = __builtin_convertvector(h, v8bf);
    }
    m2 = wave_max(m2);

    float s2 = 0.f;
    for (int i = lane * 8; i < MEM_DIM; i += 256) {
      v8f x = __builtin_convertvector(*(const v8bf*)(row + i), v8f);
#pragma unroll
      for (int j = 0; j < 8; ++j) s2 += __expf(x[j] - m2);
    }
    s2 = wave_sum(s2);
    const float inv2 = 1.0f / s2;

    float* gout = att_out + (size_t)(rb + r) * MEM_DIM;
    for (int i = lane * 8; i < MEM_DIM; i += 256) {
      v8f x = __builtin_convertvector(*(const v8bf*)(row + i), v8f);
      v8f a;
#pragma unroll
      for (int j = 0; j < 8; ++j) a[j] = __expf(x[j] - m2) * inv2;  // softmax2
      *(v8bf*)(row + i) = __builtin_convertvector(a, v8bf); // A-operand (GEMM2)
      *(v8f*)(gout + i) = a;                                // f32 out, 32B/lane
    }
  }
  __syncthreads();

  // ---- GEMM2: z_hat[32 x 1024] = att(bf16 in LDS) @ mem (bf16 via mem^T) ----
  {
    const bf16_t* arow0 = wbf + ln * MEM_DIM;
    const bf16_t* arow1 = wbf + (16 + ln) * MEM_DIM;
    for (int nt = wid; nt < FEAT / 16; nt += NWAVES) {
      const int nbase = nt * 16;
      const bf16_t* brow = memT + (size_t)(nbase + ln) * MEM_DIM;
      v8f acc0 = {0.f, 0.f, 0.f, 0.f, 0.f, 0.f, 0.f, 0.f};
      v8f acc1 = {0.f, 0.f, 0.f, 0.f, 0.f, 0.f, 0.f, 0.f};
#pragma unroll 4
      for (int k = 0; k < MEM_DIM; k += 32) {
        v16bf b  = load_afrag(brow + k + koffA);
        v16bf a0 = load_afrag(arow0 + k + koffA);
        v16bf a1 = load_afrag(arow1 + k + koffA);
        __builtin_prefetch(brow + k + 256, 0, 3);
        acc0 = __builtin_amdgcn_wmma_f32_16x16x32_bf16(false, a0, false, b,
                                                       (short)0, acc0, false, false);
        acc1 = __builtin_amdgcn_wmma_f32_16x16x32_bf16(false, a1, false, b,
                                                       (short)0, acc1, false, false);
      }
#pragma unroll
      for (int v = 0; v < 8; ++v) {
        zhat_out[(size_t)(rb + v + 8 * half) * FEAT + nbase + ln]      = acc0[v];
        zhat_out[(size_t)(rb + 16 + v + 8 * half) * FEAT + nbase + ln] = acc1[v];
      }
    }
  }
}

extern "C" void kernel_launch(void* const* d_in, const int* in_sizes, int n_in,
                              void* d_out, int out_size, void* d_ws, size_t ws_size,
                              hipStream_t stream) {
  const float* z   = (const float*)d_in[0];  // [16384, 1024]
  const float* mem = (const float*)d_in[1];  // [4096, 1024]

  bf16_t* memB = (bf16_t*)d_ws;                                  // 8 MB
  bf16_t* memT = memB + (size_t)MEM_DIM * FEAT;                  // 8 MB

  float* att_out  = (float*)d_out;                               // [16384, 4096]
  float* zhat_out = att_out + (size_t)BATCH * MEM_DIM;           // [16384, 1024]

  hipFuncSetAttribute((const void*)mem_attn_kernel,
                      hipFuncAttributeMaxDynamicSharedMemorySize, SMEM_BYTES);

  prep_mem_kernel<<<MEM_DIM, 256, 0, stream>>>(mem, memB, memT);
  mem_attn_kernel<<<BATCH / TILE_B, THREADS, SMEM_BYTES, stream>>>(
      z, memB, memT, att_out, zhat_out);
}
